// KBLaMBitNetAttention_18047452578030
// MI455X (gfx1250) — 16-bit and 8-bit matrix operand order repaired and checked by reading, compiled, not run on hardware
//
#include <hip/hip_runtime.h>
#include <hip/hip_bf16.h>
#include <stdint.h>

#define QLEN  1024
#define HDIM  2048
#define NH    32
#define NKV   8
#define HD    64
#define KBLEN 2048
#define GQA   (NH / NKV)
#define SCALE 0.125f
#define KB_BIAS 0.69314718f

typedef int    v8i   __attribute__((ext_vector_type(8)));
typedef float  v8f   __attribute__((ext_vector_type(8)));
typedef __bf16 v16bf __attribute__((ext_vector_type(16)));
typedef __bf16 v8bf  __attribute__((ext_vector_type(8)));

__global__ void zero_kernel(float* p, int n) {
    int i = blockIdx.x * blockDim.x + threadIdx.x;
    if (i < n) p[i] = 0.f;
}

__global__ __launch_bounds__(256)
void absum_kernel(const float* __restrict__ w, float* __restrict__ acc, int n) {
    __shared__ float red[256];
    float s = 0.f;
    for (int i = blockIdx.x * 256 + threadIdx.x; i < n; i += gridDim.x * 256)
        s += fabsf(w[i]);
    red[threadIdx.x] = s;
    __syncthreads();
    for (int st = 128; st > 0; st >>= 1) {
        if (threadIdx.x < st) red[threadIdx.x] += red[threadIdx.x + st];
        __syncthreads();
    }
    if (threadIdx.x == 0) atomicAdd(acc, red[0]);
}

__global__ __launch_bounds__(256)
void wquant_kernel(const float* __restrict__ w, int8_t* __restrict__ w8,
                   const float* __restrict__ acc, float cntInv, int n) {
    int i = blockIdx.x * 256 + threadIdx.x;
    if (i >= n) return;
    float mean = fmaxf(acc[0] * cntInv, 1e-5f);
    float q = rintf(w[i] / mean);
    q = fminf(fmaxf(q, -1.f), 1.f);
    w8[i] = (int8_t)q;
}

__global__ __launch_bounds__(256)
void aquant_kernel(const float* __restrict__ x, int8_t* __restrict__ x8,
                   float* __restrict__ aInv, int cols) {
    __shared__ float red[256];
    const int row = blockIdx.x;
    const float* xr = x + (size_t)row * cols;
    float mx = 0.f;
    for (int c = threadIdx.x; c < cols; c += 256) mx = fmaxf(mx, fabsf(xr[c]));
    red[threadIdx.x] = mx;
    __syncthreads();
    for (int st = 128; st > 0; st >>= 1) {
        if (threadIdx.x < st) red[threadIdx.x] = fmaxf(red[threadIdx.x], red[threadIdx.x + st]);
        __syncthreads();
    }
    const float amax = fmaxf(red[0], 1e-5f);
    const float sc = 127.f / amax;
    int8_t* qr = x8 + (size_t)row * cols;
    for (int c = threadIdx.x; c < cols; c += 256) {
        float q = rintf(xr[c] * sc);
        q = fminf(fmaxf(q, -128.f), 127.f);
        qr[c] = (int8_t)q;
    }
    if (threadIdx.x == 0) aInv[row] = amax / 127.f;
}

__global__ __launch_bounds__(256)
void gemm_i8_kernel(const int8_t* __restrict__ A8, const float* __restrict__ aInv,
                    const int8_t* __restrict__ B8, const float* __restrict__ wAcc,
                    float wCntInv, float* __restrict__ C, int M, int N, int K) {
    const int lane = threadIdx.x & 31;
    const int wv   = threadIdx.x >> 5;
    const int m    = lane & 15;
    const int half = lane >> 4;
    const int mBase = blockIdx.y * 16;
    const int nBase = (blockIdx.x * 8 + wv) * 64;
    if (mBase >= M || nBase >= N) return;

    const int8_t* aRow = A8 + (size_t)(mBase + m) * K;
    const int8_t* bRow[4];
#pragma unroll
    for (int nt = 0; nt < 4; ++nt)
        bRow[nt] = B8 + (size_t)(nBase + nt * 16 + m) * K;

    v8i acc[4];
#pragma unroll
    for (int nt = 0; nt < 4; ++nt)
#pragma unroll
        for (int r = 0; r < 8; ++r) acc[nt][r] = 0;

    for (int k0 = 0; k0 < K; k0 += 64) {
        __builtin_prefetch(aRow + k0 + 128, 0, 3);
        __builtin_prefetch(bRow[0] + k0 + 128, 0, 3);
        v8i a;
#pragma unroll
        for (int v = 0; v < 8; ++v) {
            const int ka = (v >> 1) * 16 + (v & 1) * 4 + half * 8;
            a[v] = *(const int*)(aRow + k0 + ka);
        }
#pragma unroll
        for (int nt = 0; nt < 4; ++nt) {
            v8i b;
#pragma unroll
            for (int v = 0; v < 8; ++v) {
                const int kb = (v >> 1) * 16 + (v & 1) * 4 + half * 8;
                b[v] = *(const int*)(bRow[nt] + k0 + kb);
            }
            acc[nt] = __builtin_amdgcn_wmma_i32_16x16x64_iu8(true, a, true, b,
                                                             acc[nt], false, false);
        }
    }

    const float wInv = fmaxf(wAcc[0] * wCntInv, 1e-5f);
#pragma unroll
    for (int nt = 0; nt < 4; ++nt)
#pragma unroll
        for (int r = 0; r < 8; ++r) {
            const int row = mBase + r + 8 * half;
            C[(size_t)row * N + nBase + nt * 16 + m] = (float)acc[nt][r] * aInv[row] * wInv;
        }
}

__global__ void rope_cvt_kernel(const float* __restrict__ xin, __bf16* __restrict__ out,
                                const int* __restrict__ pos, int nHeads) {
    const int idx = blockIdx.x * 256 + threadIdx.x;
    const int total = QLEN * nHeads * HD;
    if (idx >= total) return;
    const int d = idx % HD;
    const int h = (idx / HD) % nHeads;
    const int p = idx / (HD * nHeads);
    const float* row = xin + (size_t)p * nHeads * HD + h * HD;
    const float v = row[d];
    const float rot = (d < 32) ? -row[d + 32] : row[d - 32];
    const int i = d & 31;
    const float invf = __powf(10000.f, -(float)(2 * i) * (1.f / 64.f));
    const float fr = (float)pos[p] * invf;
    out[((size_t)h * QLEN + p) * HD + d] = (__bf16)(v * __cosf(fr) + rot * __sinf(fr));
}

__global__ void cvt_heads_kernel(const float* __restrict__ xin, __bf16* __restrict__ out,
                                 int nHeads) {
    const int idx = blockIdx.x * 256 + threadIdx.x;
    const int total = QLEN * nHeads * HD;
    if (idx >= total) return;
    const int d = idx % HD;
    const int h = (idx / HD) % nHeads;
    const int p = idx / (HD * nHeads);
    out[((size_t)h * QLEN + p) * HD + d] = (__bf16)xin[(size_t)p * nHeads * HD + h * HD + d];
}

__global__ void cvt_flat_kernel(const float* __restrict__ in, __bf16* __restrict__ out, int n) {
    const int i = blockIdx.x * 256 + threadIdx.x;
    if (i < n) out[i] = (__bf16)in[i];
}

__global__ void transpose_v_kernel(const float* __restrict__ in, __bf16* __restrict__ out,
                                   int nHeads) {
    const int idx = blockIdx.x * 256 + threadIdx.x;
    const int total = QLEN * nHeads * HD;
    if (idx >= total) return;
    const int d = idx % HD;
    const int h = (idx / HD) % nHeads;
    const int p = idx / (HD * nHeads);
    out[((size_t)h * HD + d) * QLEN + p] = (__bf16)in[(size_t)p * nHeads * HD + h * HD + d];
}

__global__ void transpose_kbv_kernel(const float* __restrict__ in, __bf16* __restrict__ out) {
    const int idx = blockIdx.x * 256 + threadIdx.x;
    const int total = NH * KBLEN * HD;
    if (idx >= total) return;
    const int d = idx % HD;
    const int kp = (idx / HD) % KBLEN;
    const int h = idx / (HD * KBLEN);
    out[((size_t)h * HD + d) * KBLEN + kp] = (__bf16)in[idx];
}

__device__ __forceinline__ void load_afrag(const __bf16* row, int dBase, int half, v16bf& a) {
    v8bf lo = *(const v8bf*)(row + dBase + half * 16);
    v8bf hi = *(const v8bf*)(row + dBase + half * 16 + 8);
#pragma unroll
    for (int i = 0; i < 8; ++i) { a[i] = lo[i]; a[8 + i] = hi[i]; }
}

__global__ __launch_bounds__(32)
void flash_kernel(const __bf16* __restrict__ qr,
                  const __bf16* __restrict__ kbq,
                  const __bf16* __restrict__ kr,
                  const __bf16* __restrict__ vt,
                  const __bf16* __restrict__ kbk,
                  const __bf16* __restrict__ kbvt,
                  float* __restrict__ attnO)
{
    __shared__ __bf16 Pl[16 * 32];
    const int h = blockIdx.y;
    const int qBase = blockIdx.x * 16;
    const int lane = threadIdx.x;
    const int m = lane & 15;
    const int half = lane >> 4;
    const int kvh = h / GQA;

    v16bf aq[2], akb[2];
    {
        const __bf16* qrow = qr  + ((size_t)h * QLEN + qBase + m) * HD;
        const __bf16* brow = kbq + ((size_t)h * QLEN + qBase + m) * HD;
#pragma unroll
        for (int dc = 0; dc < 2; ++dc) {
            load_afrag(qrow, dc * 32, half, aq[dc]);
            load_afrag(brow, dc * 32, half, akb[dc]);
        }
    }

    float mi[8], li[8];
    v8f O[4];
#pragma unroll
    for (int r = 0; r < 8; ++r) { mi[r] = -1e30f; li[r] = 0.f; }
#pragma unroll
    for (int nt = 0; nt < 4; ++nt)
#pragma unroll
        for (int r = 0; r < 8; ++r) O[nt][r] = 0.f;

#pragma unroll 1
    for (int phase = 0; phase < 2; ++phase) {
        const bool isKB = (phase == 0);
        const __bf16* keys = isKB ? (kbk + (size_t)h * KBLEN * HD)
                                  : (kr + (size_t)kvh * QLEN * HD);
        const __bf16* vals = isKB ? (kbvt + (size_t)h * HD * KBLEN)
                                  : (vt + (size_t)kvh * HD * QLEN);
        const int vstride = isKB ? KBLEN : QLEN;
        const int kEnd = isKB ? KBLEN : (qBase + 16);
        const float bias = isKB ? KB_BIAS : 0.f;
        const v16bf* A = isKB ? akb : aq;

        for (int kc = 0; kc < kEnd; kc += 32) {
            v8f S0 = {0, 0, 0, 0, 0, 0, 0, 0};
            v8f S1 = {0, 0, 0, 0, 0, 0, 0, 0};
            const __bf16* kr0 = keys + (size_t)(kc + m) * HD;
            const __bf16* kr1 = keys + (size_t)(kc + 16 + m) * HD;
#pragma unroll
            for (int dc = 0; dc < 2; ++dc) {
                v16bf b0 = *(const v16bf*)(kr0 + dc * 32 + half * 16);
                v16bf b1 = *(const v16bf*)(kr1 + dc * 32 + half * 16);
                S0 = __builtin_amdgcn_wmma_f32_16x16x32_bf16(false, A[dc], false, b0,
                                                             (short)0, S0, false, false);
                S1 = __builtin_amdgcn_wmma_f32_16x16x32_bf16(false, A[dc], false, b1,
                                                             (short)0, S1, false, false);
            }

            float s0[8], s1[8];
#pragma unroll
            for (int r = 0; r < 8; ++r) {
                s0[r] = S0[r] * SCALE + bias;
                s1[r] = S1[r] * SCALE + bias;
                if (!isKB) {
                    const int row = qBase + r + 8 * half;
                    if (kc + m > row)      s0[r] = -1e30f;
                    if (kc + 16 + m > row) s1[r] = -1e30f;
                }
            }

#pragma unroll
            for (int r = 0; r < 8; ++r) {
                float cm = fmaxf(s0[r], s1[r]);
#pragma unroll
                for (int sft = 1; sft < 16; sft <<= 1)
                    cm = fmaxf(cm, __shfl_xor(cm, sft, 32));
                const float mn = fmaxf(mi[r], cm);
                const float fac = __expf(mi[r] - mn);
                const float p0 = __expf(s0[r] - mn);
                const float p1 = __expf(s1[r] - mn);
                float ps = p0 + p1;
#pragma unroll
                for (int sft = 1; sft < 16; sft <<= 1)
                    ps += __shfl_xor(ps, sft, 32);
                li[r] = li[r] * fac + ps;
                mi[r] = mn;
#pragma unroll
                for (int nt = 0; nt < 4; ++nt) O[nt][r] *= fac;
                const int row = r + 8 * half;
                Pl[row * 32 + m] = (__bf16)p0;
                Pl[row * 32 + 16 + m] = (__bf16)p1;
            }
            __syncthreads();

            v16bf Pa;
            {
                v8bf lo = *(const v8bf*)(&Pl[m * 32 + half * 16]);
                v8bf hi = *(const v8bf*)(&Pl[m * 32 + half * 16 + 8]);
#pragma unroll
                for (int i = 0; i < 8; ++i) { Pa[i] = lo[i]; Pa[8 + i] = hi[i]; }
            }

#pragma unroll
            for (int nt = 0; nt < 4; ++nt) {
                const __bf16* vrow = vals + (size_t)(nt * 16 + m) * vstride + kc + half * 16;
                v16bf bv = *(const v16bf*)vrow;
                O[nt] = __builtin_amdgcn_wmma_f32_16x16x32_bf16(false, Pa, false, bv,
                                                                (short)0, O[nt], false, false);
            }
            __syncthreads();
        }
    }

#pragma unroll
    for (int nt = 0; nt < 4; ++nt)
#pragma unroll
        for (int r = 0; r < 8; ++r) {
            const int row = qBase + r + 8 * half;
            attnO[(size_t)row * HDIM + h * HD + nt * 16 + m] = O[nt][r] / li[r];
        }
}

extern "C" void kernel_launch(void* const* d_in, const int* in_sizes, int n_in,
                              void* d_out, int out_size, void* d_ws, size_t ws_size,
                              hipStream_t stream) {
    const float* hidden   = (const float*)d_in[0];
    const float* kb_keys  = (const float*)d_in[2];
    const float* kb_vals  = (const float*)d_in[3];
    const float* Wq       = (const float*)d_in[4];
    const float* Wk       = (const float*)d_in[5];
    const float* Wv       = (const float*)d_in[6];
    const float* Wo       = (const float*)d_in[7];
    const float* Wq_new   = (const float*)d_in[8];
    const int*   pos      = (const int*)d_in[9];
    float* out = (float*)d_out;

    char* p = (char*)d_ws;
    auto alloc = [&](size_t bytes) -> char* {
        char* r = p;
        p += (bytes + 255) & ~(size_t)255;
        return r;
    };
    float*  wsum  = (float*)alloc(8 * sizeof(float));
    int8_t* Wq8   = (int8_t*)alloc((size_t)HDIM * HDIM);
    int8_t* Wk8   = (int8_t*)alloc((size_t)NKV * HD * HDIM);
    int8_t* Wv8   = (int8_t*)alloc((size_t)NKV * HD * HDIM);
    int8_t* Wqn8  = (int8_t*)alloc((size_t)HDIM * HDIM);
    int8_t* Wo8   = (int8_t*)alloc((size_t)HDIM * HDIM);
    int8_t* x8    = (int8_t*)alloc((size_t)QLEN * HDIM);
    float*  xaInv = (float*)alloc(QLEN * sizeof(float));
    int8_t* o8    = (int8_t*)alloc((size_t)QLEN * HDIM);
    float*  oaInv = (float*)alloc(QLEN * sizeof(float));
    float*  qf    = (float*)alloc((size_t)QLEN * HDIM * 4);
    float*  kf    = (float*)alloc((size_t)QLEN * NKV * HD * 4);
    float*  vf    = (float*)alloc((size_t)QLEN * NKV * HD * 4);
    float*  kbqf  = (float*)alloc((size_t)QLEN * HDIM * 4);
    float*  attnO = (float*)alloc((size_t)QLEN * HDIM * 4);
    __bf16* qr    = (__bf16*)alloc((size_t)NH * QLEN * HD * 2);
    __bf16* kbqb  = (__bf16*)alloc((size_t)NH * QLEN * HD * 2);
    __bf16* krb   = (__bf16*)alloc((size_t)NKV * QLEN * HD * 2);
    __bf16* vtb   = (__bf16*)alloc((size_t)NKV * HD * QLEN * 2);
    __bf16* kbkb  = (__bf16*)alloc((size_t)NH * KBLEN * HD * 2);
    __bf16* kbvtb = (__bf16*)alloc((size_t)NH * HD * KBLEN * 2);

    const int nWq = HDIM * HDIM, nWk = NKV * HD * HDIM;

    zero_kernel<<<1, 32, 0, stream>>>(wsum, 8);
    absum_kernel<<<512, 256, 0, stream>>>(Wq,     wsum + 0, nWq);
    absum_kernel<<<512, 256, 0, stream>>>(Wk,     wsum + 1, nWk);
    absum_kernel<<<512, 256, 0, stream>>>(Wv,     wsum + 2, nWk);
    absum_kernel<<<512, 256, 0, stream>>>(Wq_new, wsum + 3, nWq);
    absum_kernel<<<512, 256, 0, stream>>>(Wo,     wsum + 4, nWq);
    wquant_kernel<<<(nWq + 255) / 256, 256, 0, stream>>>(Wq,     Wq8,  wsum + 0, 1.f / nWq, nWq);
    wquant_kernel<<<(nWk + 255) / 256, 256, 0, stream>>>(Wk,     Wk8,  wsum + 1, 1.f / nWk, nWk);
    wquant_kernel<<<(nWk + 255) / 256, 256, 0, stream>>>(Wv,     Wv8,  wsum + 2, 1.f / nWk, nWk);
    wquant_kernel<<<(nWq + 255) / 256, 256, 0, stream>>>(Wq_new, Wqn8, wsum + 3, 1.f / nWq, nWq);
    wquant_kernel<<<(nWq + 255) / 256, 256, 0, stream>>>(Wo,     Wo8,  wsum + 4, 1.f / nWq, nWq);

    aquant_kernel<<<QLEN, 256, 0, stream>>>(hidden, x8, xaInv, HDIM);
    gemm_i8_kernel<<<dim3(HDIM / 512, QLEN / 16), 256, 0, stream>>>(
        x8, xaInv, Wq8, wsum + 0, 1.f / nWq, qf, QLEN, HDIM, HDIM);
    gemm_i8_kernel<<<dim3((NKV * HD) / 512, QLEN / 16), 256, 0, stream>>>(
        x8, xaInv, Wk8, wsum + 1, 1.f / nWk, kf, QLEN, NKV * HD, HDIM);
    gemm_i8_kernel<<<dim3((NKV * HD) / 512, QLEN / 16), 256, 0, stream>>>(
        x8, xaInv, Wv8, wsum + 2, 1.f / nWk, vf, QLEN, NKV * HD, HDIM);
    gemm_i8_kernel<<<dim3(HDIM / 512, QLEN / 16), 256, 0, stream>>>(
        x8, xaInv, Wqn8, wsum + 3, 1.f / nWq, kbqf, QLEN, HDIM, HDIM);

    const int nQ = QLEN * NH * HD, nK = QLEN * NKV * HD, nKB = NH * KBLEN * HD;
    rope_cvt_kernel<<<(nQ + 255) / 256, 256, 0, stream>>>(qf, qr, pos, NH);
    rope_cvt_kernel<<<(nK + 255) / 256, 256, 0, stream>>>(kf, krb, pos, NKV);
    transpose_v_kernel<<<(nK + 255) / 256, 256, 0, stream>>>(vf, vtb, NKV);
    cvt_heads_kernel<<<(nQ + 255) / 256, 256, 0, stream>>>(kbqf, kbqb, NH);
    cvt_flat_kernel<<<(nKB + 255) / 256, 256, 0, stream>>>(kb_keys, kbkb, nKB);
    transpose_kbv_kernel<<<(nKB + 255) / 256, 256, 0, stream>>>(kb_vals, kbvtb);

    flash_kernel<<<dim3(QLEN / 16, NH), 32, 0, stream>>>(qr, kbqb, krb, vtb, kbkb, kbvtb, attnO);

    aquant_kernel<<<QLEN, 256, 0, stream>>>(attnO, o8, oaInv, HDIM);
    gemm_i8_kernel<<<dim3(HDIM / 512, QLEN / 16), 256, 0, stream>>>(
        o8, oaInv, Wo8, wsum + 4, 1.f / nWq, out, QLEN, HDIM, HDIM);
}
